// GraphVAE_51642686767340
// MI455X (gfx1250) — compile-verified, matrix operand order checked by reading, not run on hardware
//
#include <hip/hip_runtime.h>
#include <math.h>

#define N_NODES 50000
#define N_EDGES 800000
#define FEAT    128
#define HID     64
#define LAT     32
#define NG      64
#define MN      30

typedef float v2f  __attribute__((ext_vector_type(2)));
typedef float v8f  __attribute__((ext_vector_type(8)));
typedef _Float16 v16h __attribute__((ext_vector_type(16)));

// ---------------------------------------------------------------- utilities
__global__ void zero_k(float* __restrict__ p, int n) {
    int t = blockIdx.x * blockDim.x + threadIdx.x;
    if (t < n) p[t] = 0.0f;
}

__global__ void degree_k(const int* __restrict__ dst, float* __restrict__ deg, int E) {
    int t = blockIdx.x * blockDim.x + threadIdx.x;
    if (t < E) atomicAdd(&deg[dst[t]], 1.0f);
}

__global__ void rsqrt_k(float* __restrict__ dinv, int n) {
    int t = blockIdx.x * blockDim.x + threadIdx.x;
    if (t < n) dinv[t] = rsqrtf(dinv[t] + 1.0f);   // deg includes +1 self term
}

__global__ void enorm_k(const int* __restrict__ src, const int* __restrict__ dst,
                        const float* __restrict__ dinv, float* __restrict__ enorm, int E) {
    int t = blockIdx.x * blockDim.x + threadIdx.x;
    if (t < E) enorm[t] = dinv[src[t]] * dinv[dst[t]];
}

// ---------------------------------------------------------------- WMMA GEMM
// C[M x N] = A[M x K] * B[K x N], row-major, fp32. N == 64 (4 n-tiles/block,
// threadIdx.y picks the n-tile). M must be a multiple of 16 (50000 = 3125*16).
__global__ __launch_bounds__(128)
void gemm_wmma(const float* __restrict__ A, const float* __restrict__ B,
               float* __restrict__ C, int M, int K, int N) {
    const int lane = threadIdx.x;        // 0..31
    const int half = lane >> 4;          // 0/1
    const int mn   = lane & 15;
    const int m    = blockIdx.x * 16 + mn;
    const int n    = threadIdx.y * 16 + mn;
    const float* arow = A + (size_t)m * K;
    v8f acc = {};
#if __has_builtin(__builtin_amdgcn_wmma_f32_16x16x4_f32)
    // full-precision path: V_WMMA_F32_16X16X4_F32
    for (int k0 = 0; k0 < K; k0 += 4) {
        const int k = k0 + 2 * half;               // A: VGPR v holds K = k + v
        v2f a; a.x = arow[k]; a.y = arow[k + 1];
        v2f b; b.x = B[(size_t)k * N + n]; b.y = B[(size_t)(k + 1) * N + n];
        acc = __builtin_amdgcn_wmma_f32_16x16x4_f32(
                  false, a, false, b, (short)0, acc, false, false);
    }
#else
    // fallback: convert to f16, V_WMMA_F32_16X16X32_F16 (probe-verified)
    for (int k0 = 0; k0 < K; k0 += 32) {
        v16h a, b;
        #pragma unroll
        for (int i = 0; i < 16; ++i) {
            const int v = i >> 1, p = i & 1;
            const int k = k0 + ((v & 3) * 2 + p) + half * 8 + ((v >> 2) * 16);
            a[i] = (_Float16)arow[k];
            b[i] = (_Float16)B[(size_t)k * N + n];
        }
        acc = __builtin_amdgcn_wmma_f32_16x16x32_f16(
                  false, a, false, b, (short)0, acc, false, false);
    }
#endif
    // C/D layout: VGPR v -> row (v + 8*half), col n
    float* crow = C + (size_t)(blockIdx.x * 16 + 8 * half) * N + n;
    #pragma unroll
    for (int v = 0; v < 8; ++v) crow[(size_t)v * N] = acc[v];
}

// ---------------------------------------------------------------- edge scatter
// agg[dst, :] += hW[src, :] * enorm[e];   16 threads per edge, 4 feats each.
__global__ __launch_bounds__(256)
void scatter_edges(const float* __restrict__ hw, const int* __restrict__ src,
                   const int* __restrict__ dst, const float* __restrict__ enorm,
                   float* __restrict__ agg, int E) {
    int t = blockIdx.x * blockDim.x + threadIdx.x;
    if (t >= E * 16) return;
    const int e = t >> 4;
    const int f = (t & 15) << 2;
    const int s = src[e], d = dst[e];
    const float en = enorm[e];
    const float4 v = *(const float4*)(hw + (size_t)s * HID + f);
    float* ap = agg + (size_t)d * HID + f;
    atomicAdd(ap + 0, v.x * en);
    atomicAdd(ap + 1, v.y * en);
    atomicAdd(ap + 2, v.z * en);
    atomicAdd(ap + 3, v.w * en);
}

// h_out = relu(agg + hW * dinv^2 + bias)
__global__ void combine_gcn(const float* __restrict__ agg, const float* __restrict__ hw,
                            const float* __restrict__ dinv, const float* __restrict__ bias,
                            float* __restrict__ out, int n) {
    int t = blockIdx.x * blockDim.x + threadIdx.x;
    if (t >= n * HID) return;
    const int i = t >> 6, f = t & 63;
    const float di = dinv[i];
    float v = agg[t] + hw[t] * di * di + bias[f];
    out[t] = v > 0.0f ? v : 0.0f;
}

// ---------------------------------------------------------------- pooling
__global__ void pool_counts_k(const int* __restrict__ batch, float* __restrict__ counts, int n) {
    int t = blockIdx.x * blockDim.x + threadIdx.x;
    if (t < n) atomicAdd(&counts[batch[t]], 1.0f);
}

__global__ void pool_sum_k(const float* __restrict__ h, const int* __restrict__ batch,
                           float* __restrict__ pooled, int n) {
    int t = blockIdx.x * blockDim.x + threadIdx.x;
    if (t >= n * HID) return;
    const int i = t >> 6, f = t & 63;
    atomicAdd(&pooled[(size_t)batch[i] * HID + f], h[t]);
}

// ---------------------------------------------------------------- tiny tail
// pooled -> mu/logvar ; z=mu[0] -> decoder -> 30x30 adjacency. One block.
__global__ __launch_bounds__(256)
void tail_kernel(const float* __restrict__ pooled, const float* __restrict__ counts,
                 const float* __restrict__ Wmu, const float* __restrict__ bmu,
                 const float* __restrict__ Wlv, const float* __restrict__ blv,
                 const float* __restrict__ Wl1, const float* __restrict__ bl1,
                 const float* __restrict__ Wl2, const float* __restrict__ bl2,
                 const float* __restrict__ We1, const float* __restrict__ be1,
                 const float* __restrict__ We2, const float* __restrict__ be2,
                 float* __restrict__ out) {
    __shared__ float sp[NG * HID];     // pooled / counts
    __shared__ float sz[LAT];          // z = mu[graph 0]
    __shared__ float st[HID];          // relu(z @ Wl1 + bl1)
    __shared__ float semb[MN * HID];   // node_emb[0, :30]
    __shared__ float shi[MN * HID];
    __shared__ float shj[MN * HID];
    const int tid = threadIdx.x;

    for (int i = tid; i < NG * HID; i += 256) {
        float c = counts[i >> 6];
        c = c > 1.0f ? c : 1.0f;
        sp[i] = pooled[i] / c;
    }
    __syncthreads();

    // mu (out[900..2947]) and logvar (out[2948..4995])
    for (int i = tid; i < NG * LAT; i += 256) {
        const int g = i >> 5, j = i & 31;
        float smu = bmu[j], slv = blv[j];
        const float* pg = sp + g * HID;
        for (int k = 0; k < HID; ++k) {
            const float p = pg[k];
            smu += p * Wmu[k * LAT + j];
            slv += p * Wlv[k * LAT + j];
        }
        out[900 + i]        = smu;
        out[900 + 2048 + i] = slv;
        if (g == 0) sz[j] = smu;        // z = mu, only graph 0 used below
    }
    __syncthreads();

    if (tid < HID) {
        float s = bl1[tid];
        for (int j = 0; j < LAT; ++j) s += sz[j] * Wl1[j * HID + tid];
        st[tid] = s > 0.0f ? s : 0.0f;
    }
    __syncthreads();

    for (int i = tid; i < MN * HID; i += 256) {      // emb = t @ Wl2 + bl2
        float s = bl2[i];
        for (int j = 0; j < HID; ++j) s += st[j] * Wl2[(size_t)j * (HID * MN) + i];
        semb[i] = s;
    }
    __syncthreads();

    for (int i = tid; i < MN * HID; i += 256) {      // hi = emb@Wtop, hj = emb@Wbot
        const int n = i >> 6, k = i & 63;
        const float* e = semb + n * HID;
        float a = 0.0f, b = 0.0f;
        for (int h = 0; h < HID; ++h) {
            const float eh = e[h];
            a += eh * We1[h * HID + k];
            b += eh * We1[(HID + h) * HID + k];
        }
        shi[i] = a; shj[i] = b;
    }
    __syncthreads();

    for (int i = tid; i < MN * MN; i += 256) {       // adj = triu(probs,1) symmetrized
        const int r = i / MN, c = i % MN;
        float val = 0.0f;
        if (r != c) {
            const int a = r < c ? r : c;
            const int b = r < c ? c : r;
            float s = be2[0];
            const float* hia = shi + a * HID;
            const float* hjb = shj + b * HID;
            for (int k = 0; k < HID; ++k) {
                float u = hia[k] + hjb[k] + be1[k];
                u = u > 0.0f ? u : 0.0f;
                s += u * We2[k];
            }
            val = 1.0f / (1.0f + expf(-s));
        }
        out[i] = val;
    }
}

// ---------------------------------------------------------------- launcher
extern "C" void kernel_launch(void* const* d_in, const int* in_sizes, int n_in,
                              void* d_out, int out_size, void* d_ws, size_t ws_size,
                              hipStream_t stream) {
    (void)in_sizes; (void)n_in; (void)out_size; (void)ws_size;
    const float* x     = (const float*)d_in[0];
    const int*   ei    = (const int*)d_in[1];      // (2, E): src row then dst row
    const int*   batch = (const int*)d_in[2];
    const float* W1  = (const float*)d_in[4],  *b1  = (const float*)d_in[5];
    const float* W2  = (const float*)d_in[6],  *b2  = (const float*)d_in[7];
    const float* Wmu = (const float*)d_in[8],  *bmu = (const float*)d_in[9];
    const float* Wlv = (const float*)d_in[10], *blv = (const float*)d_in[11];
    const float* Wl1 = (const float*)d_in[12], *bl1 = (const float*)d_in[13];
    const float* Wl2 = (const float*)d_in[14], *bl2 = (const float*)d_in[15];
    const float* We1 = (const float*)d_in[16], *be1 = (const float*)d_in[17];
    const float* We2 = (const float*)d_in[18], *be2 = (const float*)d_in[19];
    float* out = (float*)d_out;
    float* ws  = (float*)d_ws;

    // workspace carve-up (floats)
    float* dinv   = ws;                               // N
    float* enorm  = ws + 50048;                       // E
    float* hW     = enorm + N_EDGES;                  // N*64
    float* agg    = hW  + (size_t)N_NODES * HID;      // N*64
    float* h      = agg + (size_t)N_NODES * HID;      // N*64
    float* pooled = h   + (size_t)N_NODES * HID;      // 64*64
    float* counts = pooled + NG * HID;                // 64

    const int* src = ei;
    const int* dst = ei + N_EDGES;
    const int NH = N_NODES * HID;
    auto cdiv = [](int a, int b) { return (a + b - 1) / b; };

    // normalization terms
    zero_k  <<<cdiv(N_NODES, 256), 256, 0, stream>>>(dinv, N_NODES);
    degree_k<<<cdiv(N_EDGES, 256), 256, 0, stream>>>(dst, dinv, N_EDGES);
    rsqrt_k <<<cdiv(N_NODES, 256), 256, 0, stream>>>(dinv, N_NODES);
    enorm_k <<<cdiv(N_EDGES, 256), 256, 0, stream>>>(src, dst, dinv, enorm, N_EDGES);

    // GCN layer 1
    gemm_wmma<<<dim3(N_NODES / 16), dim3(32, 4), 0, stream>>>(x, W1, hW, N_NODES, FEAT, HID);
    zero_k<<<cdiv(NH, 256), 256, 0, stream>>>(agg, NH);
    scatter_edges<<<cdiv(N_EDGES * 16, 256), 256, 0, stream>>>(hW, src, dst, enorm, agg, N_EDGES);
    combine_gcn<<<cdiv(NH, 256), 256, 0, stream>>>(agg, hW, dinv, b1, h, N_NODES);

    // GCN layer 2
    gemm_wmma<<<dim3(N_NODES / 16), dim3(32, 4), 0, stream>>>(h, W2, hW, N_NODES, HID, HID);
    zero_k<<<cdiv(NH, 256), 256, 0, stream>>>(agg, NH);
    scatter_edges<<<cdiv(N_EDGES * 16, 256), 256, 0, stream>>>(hW, src, dst, enorm, agg, N_EDGES);
    combine_gcn<<<cdiv(NH, 256), 256, 0, stream>>>(agg, hW, dinv, b2, h, N_NODES);

    // mean pool
    zero_k<<<cdiv(NG * HID + NG, 256), 256, 0, stream>>>(pooled, NG * HID + NG);
    pool_counts_k<<<cdiv(N_NODES, 256), 256, 0, stream>>>(batch, counts, N_NODES);
    pool_sum_k<<<cdiv(NH, 256), 256, 0, stream>>>(h, batch, pooled, N_NODES);

    // heads + decoder tail
    tail_kernel<<<1, 256, 0, stream>>>(pooled, counts, Wmu, bmu, Wlv, blv,
                                       Wl1, bl1, Wl2, bl2, We1, be1, We2, be2, out);
}
